// GRU_Sub_83545703841926
// MI455X (gfx1250) — compile-verified, matrix operand order checked by reading, not run on hardware
//
#include <hip/hip_runtime.h>
#include <hip/hip_bf16.h>

typedef _Float16 v16h __attribute__((ext_vector_type(16)));
typedef _Float16 v8h  __attribute__((ext_vector_type(8)));
typedef float    v8f  __attribute__((ext_vector_type(8)));
typedef float    v4f  __attribute__((ext_vector_type(4)));

#define TT 1024
#define BB 128
#define HH 30
#define GG 90
#define LL 1000
#define ROWS (TT * BB)          // 131072
#define NT2 63                  // ceil(1008/16) N-tiles for output GEMM
#define HG_PITCH 20             // bank-conflict-free column store pitch

// Fast device helpers: single v_rcp_f32 / v_exp_f32 based math (no IEEE div).
__device__ __forceinline__ float fast_rcp(float x) {
    return __builtin_amdgcn_rcpf(x);
}
__device__ __forceinline__ float fast_sigmoid(float x) {
    return fast_rcp(1.f + __expf(-x));
}
__device__ __forceinline__ float fast_tanh(float x) {
    float e2 = __expf(-2.f * x);
    return 1.f - 2.f * e2 * fast_rcp(1.f + e2);
}

// ---------------------------------------------------------------------------
// Pre-pack w_out [1000][30] f32 -> f16 WMMA B-fragments (K=32 padded, N=1008)
// Fragment layout per tile: [32 lanes][16 halves], lane%16 = N, e <-> K:
//   K = (e>=8?16:0) + (e&7) + (lane>=16?8:0)
// ---------------------------------------------------------------------------
__global__ void wout_pack_kernel(const float* __restrict__ w_out,
                                 _Float16* __restrict__ wofrag) {
    int tile = blockIdx.x;          // 0..62
    int lane = threadIdx.x;         // 0..31
    int n = tile * 16 + (lane & 15);
    v16h frag;
#pragma unroll
    for (int e = 0; e < 16; ++e) {
        int k = ((e >= 8) ? 16 : 0) + (e & 7) + ((lane >= 16) ? 8 : 0);
        float v = (n < LL && k < HH) ? w_out[n * HH + k] : 0.f;
        frag[e] = (_Float16)v;
    }
    *(v16h*)(wofrag + (tile * 32 + lane) * 16) = frag;
}

// ---------------------------------------------------------------------------
// Phase 1: GRU scan. 8 blocks x 192 threads (6 waves). Block b owns batch
// rows [16b, 16b+16). Wave w computes gate N-tile w via WMMA each step.
// ---------------------------------------------------------------------------
__global__ void __launch_bounds__(192)
gru_scan_kernel(const float* __restrict__ x, const float* __restrict__ h0,
                const float* __restrict__ w_ih, const float* __restrict__ w_hh,
                const float* __restrict__ b_ih, const float* __restrict__ b_hh,
                _Float16* __restrict__ hseq, float* __restrict__ hn_out) {
    __shared__ float    hg[96 * HG_PITCH];   // gate pre-activations, col-major [96][20]
    __shared__ _Float16 h16[32 * 16];        // h as WMMA A-fragment (16 rows x K=32)
    __shared__ float    hf[16 * 32];         // h in f32, [b_local][32]

    const int tid  = threadIdx.x;
    const int lane = tid & 31;
    const int wv   = tid >> 5;               // 0..5
    const int blk  = blockIdx.x;             // batch tile
    const int j    = lane;                   // gate unit index (0..31) - fixed per thread

    // Build this wave's B-fragment of w_hh (tile wv: gates [16wv,16wv+16), K=30->32).
    // B[k][n] = w_hh[n][k] since hg = h @ w_hh^T.
    v16h bfrag;
    {
        int nabs = wv * 16 + (lane & 15);
#pragma unroll
        for (int e = 0; e < 16; ++e) {
            int k = ((e >= 8) ? 16 : 0) + (e & 7) + ((lane >= 16) ? 8 : 0);
            float v = (nabs < GG && k < HH) ? w_hh[nabs * HH + k] : 0.f;
            bfrag[e] = (_Float16)v;
        }
    }

    // Loop-invariant gate constants for this thread's j.
    float wir = 0.f, wiz = 0.f, win = 0.f;
    float bir = 0.f, biz = 0.f, bin = 0.f;
    float bhr = 0.f, bhz = 0.f, bhn = 0.f;
    if (j < HH) {
        wir = w_ih[j];      wiz = w_ih[HH + j];      win = w_ih[2 * HH + j];
        bir = b_ih[j];      biz = b_ih[HH + j];      bin = b_ih[2 * HH + j];
        bhr = b_hh[j];      bhz = b_hh[HH + j];      bhn = b_hh[2 * HH + j];
    }

    // Initialize h from h0 (covers all (b_local, j) pairs exactly once).
#pragma unroll
    for (int i = 0; i < 3; ++i) {
        int bl = i * 6 + wv;
        if (bl < 16) {
            float v = (j < HH) ? h0[(blk * 16 + bl) * HH + j] : 0.f;
            hf[bl * 32 + j] = v;
            int lf = bl + (((j & 15) >= 8) ? 16 : 0);
            int e  = ((j >= 16) ? 8 : 0) + (j & 7);
            h16[lf * 16 + e] = (_Float16)v;
        }
    }
    __syncthreads();

    for (int t = 0; t < TT; ++t) {
        // --- hg = h @ w_hh^T  (tile wv), one WMMA per wave ---
        {
            v16h a = *(const v16h*)(&h16[lane * 16]);
            v8f  c = {};
            c = __builtin_amdgcn_wmma_f32_16x16x32_f16(false, a, false, bfrag,
                                                       (short)0, c, false, false);
            int col = wv * 16 + (lane & 15);
            int r0  = (lane < 16) ? 0 : 8;
            float* p = &hg[col * HG_PITCH + r0];
            *(v4f*)(p)     = __builtin_shufflevector(c, c, 0, 1, 2, 3);
            *(v4f*)(p + 4) = __builtin_shufflevector(c, c, 4, 5, 6, 7);
        }
        __syncthreads();

        // --- gates + state update ---
#pragma unroll
        for (int i = 0; i < 3; ++i) {
            int bl = i * 6 + wv;
            if (bl < 16) {
                _Float16 hnew16 = (_Float16)0.f;
                if (j < HH) {
                    float xv = x[t * BB + blk * 16 + bl];   // wave-broadcast load
                    float hr = hg[j * HG_PITCH + bl] + bhr;
                    float hz = hg[(HH + j) * HG_PITCH + bl] + bhz;
                    float hn = hg[(2 * HH + j) * HG_PITCH + bl] + bhn;
                    float xr = __builtin_fmaf(xv, wir, bir);
                    float xz = __builtin_fmaf(xv, wiz, biz);
                    float xn = __builtin_fmaf(xv, win, bin);
                    float r  = fast_sigmoid(xr + hr);
                    float z  = fast_sigmoid(xz + hz);
                    float nn = fast_tanh(xn + r * hn);
                    float hp = hf[bl * 32 + j];
                    float hv = (1.f - z) * nn + z * hp;
                    hf[bl * 32 + j] = hv;
                    hnew16 = (_Float16)hv;
                }
                int row = t * BB + blk * 16 + bl;
                hseq[row * 32 + j] = hnew16;                // coalesced per wave
                int lf = bl + (((j & 15) >= 8) ? 16 : 0);
                int e  = ((j >= 16) ? 8 : 0) + (j & 7);
                h16[lf * 16 + e] = hnew16;
            }
        }
        __syncthreads();
    }

    // Final hidden state hn [1,B,H]
#pragma unroll
    for (int i = 0; i < 3; ++i) {
        int bl = i * 6 + wv;
        if (bl < 16 && j < HH)
            hn_out[(blk * 16 + bl) * HH + j] = hf[bl * 32 + j];
    }
}

// ---------------------------------------------------------------------------
// Phase 2: logits = hout @ w_out^T + b_out, then log_softmax over L=1000.
// 8192 blocks x 256 threads (8 waves). Block owns 16 rows; wave wv owns
// N-tiles {wv, wv+8, ...} (<=8). Logits stay register-resident; softmax
// stats via shfl + small LDS.
// ---------------------------------------------------------------------------
__global__ void __launch_bounds__(256)
out_proj_kernel(const _Float16* __restrict__ hseq,
                const _Float16* __restrict__ wofrag,
                const float* __restrict__ b_out, float* __restrict__ out) {
    __shared__ float wred[8][16];
    __shared__ float rowmaxs[16];
    __shared__ float rowzs[16];

    const int tid  = threadIdx.x;
    const int lane = tid & 31;
    const int wv   = tid >> 5;
    const int mt   = blockIdx.x;            // row tile, rows [16*mt, 16*mt+16)
    const int m    = lane & 15;
    const int r0   = (lane < 16) ? 0 : 8;

    // A-fragment: 16 rows x K=32 from hseq (row-major [row][32] halves).
    const _Float16* pa = hseq + (size_t)(mt * 16 + m) * 32;
    int k0 = (lane >= 16) ? 8 : 0;
    v8h lo = *(const v8h*)(pa + k0);
    v8h hi = *(const v8h*)(pa + k0 + 16);
    v16h a = __builtin_shufflevector(lo, hi, 0, 1, 2, 3, 4, 5, 6, 7,
                                     8, 9, 10, 11, 12, 13, 14, 15);

    v8f cc[8];
    int ntq[8];
    int nq = 0;
    for (int nt = wv; nt < NT2; nt += 8) {
        v16h b = *(const v16h*)(wofrag + (nt * 32 + lane) * 16);
        v8f c = {};
        c = __builtin_amdgcn_wmma_f32_16x16x32_f16(false, a, false, b,
                                                   (short)0, c, false, false);
        int col = nt * 16 + m;
        float bias = (col < LL) ? b_out[col] : 0.f;
#pragma unroll
        for (int i = 0; i < 8; ++i) c[i] += bias;
        cc[nq] = c;
        ntq[nq] = nt;
        ++nq;
    }

    // Row-wise max over this wave's columns.
    float pm[8];
#pragma unroll
    for (int i = 0; i < 8; ++i) pm[i] = -3.402823466e38f;
    for (int q = 0; q < nq; ++q) {
        bool valid = (ntq[q] * 16 + m) < LL;
        if (valid) {
#pragma unroll
            for (int i = 0; i < 8; ++i) pm[i] = fmaxf(pm[i], cc[q][i]);
        }
    }
#pragma unroll
    for (int i = 0; i < 8; ++i) {
        for (int mask = 1; mask < 16; mask <<= 1)
            pm[i] = fmaxf(pm[i], __shfl_xor(pm[i], mask, 32));
    }
    if (lane == 0 || lane == 16) {
#pragma unroll
        for (int i = 0; i < 8; ++i) wred[wv][r0 + i] = pm[i];
    }
    __syncthreads();
    if (tid < 16) {
        float v = wred[0][tid];
        for (int w = 1; w < 8; ++w) v = fmaxf(v, wred[w][tid]);
        rowmaxs[tid] = v;
    }
    __syncthreads();

    // Row-wise sum of exp(logit - max).
    float rm[8];
#pragma unroll
    for (int i = 0; i < 8; ++i) rm[i] = rowmaxs[r0 + i];
    float ps[8];
#pragma unroll
    for (int i = 0; i < 8; ++i) ps[i] = 0.f;
    for (int q = 0; q < nq; ++q) {
        bool valid = (ntq[q] * 16 + m) < LL;
        if (valid) {
#pragma unroll
            for (int i = 0; i < 8; ++i) ps[i] += __expf(cc[q][i] - rm[i]);
        }
    }
#pragma unroll
    for (int i = 0; i < 8; ++i) {
        for (int mask = 1; mask < 16; mask <<= 1)
            ps[i] += __shfl_xor(ps[i], mask, 32);
    }
    if (lane == 0 || lane == 16) {
#pragma unroll
        for (int i = 0; i < 8; ++i) wred[wv][r0 + i] = ps[i];
    }
    __syncthreads();
    if (tid < 16) {
        float s = 0.f;
        for (int w = 0; w < 8; ++w) s += wred[w][tid];
        rowzs[tid] = rowmaxs[tid] + __logf(s);
    }
    __syncthreads();

    // Write output = logit - logZ.
    float rz[8];
#pragma unroll
    for (int i = 0; i < 8; ++i) rz[i] = rowzs[r0 + i];
    for (int q = 0; q < nq; ++q) {
        int col = ntq[q] * 16 + m;
        if (col < LL) {
#pragma unroll
            for (int i = 0; i < 8; ++i)
                out[(size_t)(mt * 16 + r0 + i) * LL + col] = cc[q][i] - rz[i];
        }
    }
}

// ---------------------------------------------------------------------------
extern "C" void kernel_launch(void* const* d_in, const int* in_sizes, int n_in,
                              void* d_out, int out_size, void* d_ws, size_t ws_size,
                              hipStream_t stream) {
    const float* x     = (const float*)d_in[0];
    const float* h0    = (const float*)d_in[1];
    const float* w_ih  = (const float*)d_in[2];
    const float* w_hh  = (const float*)d_in[3];
    const float* b_ih  = (const float*)d_in[4];
    const float* b_hh  = (const float*)d_in[5];
    const float* w_out = (const float*)d_in[6];
    const float* b_out = (const float*)d_in[7];
    float* out = (float*)d_out;

    // Workspace layout: [0, 64KiB): w_out f16 B-fragments; [64KiB, +8MiB): hseq f16
    _Float16* wofrag = (_Float16*)d_ws;
    _Float16* hseq   = (_Float16*)((char*)d_ws + 65536);
    float* hn_out = out + (size_t)ROWS * LL;

    wout_pack_kernel<<<NT2, 32, 0, stream>>>(w_out, wofrag);
    gru_scan_kernel<<<8, 192, 0, stream>>>(x, h0, w_ih, w_hh, b_ih, b_hh,
                                           hseq, hn_out);
    out_proj_kernel<<<ROWS / 16, 256, 0, stream>>>(hseq, wofrag, b_out, out);
}
